// LSTM_73340861547038
// MI455X (gfx1250) — matrix operand order repaired and checked by reading, not compiled, not run on hardware
//
#include <hip/hip_runtime.h>
#include <math.h>

// ---------------- types for WMMA / TDM ----------------
typedef __bf16 bf16_t;
typedef __attribute__((ext_vector_type(16))) __bf16 v16bf;
typedef __attribute__((ext_vector_type(8)))  float  v8f;
typedef __attribute__((ext_vector_type(4)))  unsigned int u32x4;
typedef __attribute__((ext_vector_type(8)))  int i32x8;
typedef __attribute__((ext_vector_type(4)))  int i32x4;

union Frag {
    v16bf v;
    uint4 u[2]; // 32 bytes
};

__device__ __forceinline__ unsigned short f32_to_bf16(float f) {
    union { float f; unsigned int u; } v;
    v.f = f;
    unsigned int lsb = (v.u >> 16) & 1u;
    unsigned int rnd = v.u + 0x7fffu + lsb;
    return (unsigned short)(rnd >> 16);
}

// ---------------- constants ----------------
#define BSZ 128
#define EE  256
#define HH  516
#define H4  2064   // 4*H
#define HP  544    // H padded to multiple of 32
#define VV  32000
#define SXX 48
#define TDD 47     // decoder steps = SY-1

// ---------------- zero kernels ----------------
__global__ void zero_f32_kernel(float* p, long n) {
    long i = (long)blockIdx.x * blockDim.x + threadIdx.x;
    long stride = (long)gridDim.x * blockDim.x;
    for (; i < n; i += stride) p[i] = 0.0f;
}
__global__ void zero_u16_kernel(unsigned short* p, long n) {
    long i = (long)blockIdx.x * blockDim.x + threadIdx.x;
    long stride = (long)gridDim.x * blockDim.x;
    for (; i < n; i += stride) p[i] = 0;
}

// ---------------- f32 -> bf16 with K padding ----------------
// src: [N x K] row-major f32; dst: [N x Kpad] row-major bf16 (pad = 0)
__global__ void cvt_pad_kernel(const float* __restrict__ src,
                               unsigned short* __restrict__ dst,
                               int N, int K, int Kpad) {
    int idx = blockIdx.x * blockDim.x + threadIdx.x;
    int total = N * Kpad;
    if (idx >= total) return;
    int n = idx / Kpad;
    int k = idx - n * Kpad;
    dst[idx] = (k < K) ? f32_to_bf16(src[(long)n * K + k]) : (unsigned short)0;
}

// ---------------- embedding gather -> bf16 ----------------
// tok: [T x B] int, table: [Vsz x E] f32, out: [T x B x E] bf16
__global__ void embed_kernel(const int* __restrict__ tok,
                             const float* __restrict__ table,
                             unsigned short* __restrict__ out,
                             int T) {
    int idx = blockIdx.x * blockDim.x + threadIdx.x;
    int total = T * BSZ * EE;
    if (idx >= total) return;
    int e = idx % EE;
    int b = (idx / EE) % BSZ;
    int t = idx / (EE * BSZ);
    int token = tok[t * BSZ + b];
    out[idx] = f32_to_bf16(table[(long)token * EE + e]);
}

// ---------------- WMMA GEMM (per-wave): Out[128 x N] = A1*W1^T (+ A2*W2^T) ----------------
// One wave per 16-wide N strip, covering all M=128 with 8 accumulators.
__global__ __launch_bounds__(32) void gemm_nt_128_kernel(
    const unsigned short* __restrict__ A1, const unsigned short* __restrict__ W1, int K1,
    const unsigned short* __restrict__ A2, const unsigned short* __restrict__ W2, int K2,
    float* __restrict__ Out, int N)
{
    const int n0   = blockIdx.x << 4;
    const int lane = threadIdx.x;
    const int r    = lane & 15;
    const int hi   = lane >> 4;

    v8f acc[8];
#pragma unroll
    for (int i = 0; i < 8; ++i)
#pragma unroll
        for (int j = 0; j < 8; ++j) acc[i][j] = 0.0f;

    for (int pass = 0; pass < 2; ++pass) {
        const unsigned short* A = pass ? A2 : A1;
        const unsigned short* W = pass ? W2 : W1;
        const int K = pass ? K2 : K1;
        if (A == nullptr) continue;
        for (int k0 = 0; k0 < K; k0 += 32) {
            // B fragment: lane = column n0+r of W^T; chunks at k0+hi*8 and k0+16+hi*8
            Frag bfr;
            const unsigned short* wp = W + (size_t)(n0 + r) * K + k0 + hi * 8;
            bfr.u[0] = *(const uint4*)(wp);
            bfr.u[1] = *(const uint4*)(wp + 16);
#pragma unroll
            for (int mt = 0; mt < 8; ++mt) {
                // A fragment: lane = row mt*16+r; chunks at k0+hi*8 and k0+16+hi*8
                Frag afr;
                const unsigned short* ap = A + (size_t)(mt * 16 + r) * K + k0 + hi * 8;
                afr.u[0] = *(const uint4*)(ap);
                afr.u[1] = *(const uint4*)(ap + 16);
                acc[mt] = __builtin_amdgcn_wmma_f32_16x16x32_bf16(
                    false, afr.v, false, bfr.v, (short)0, acc[mt], false, false);
            }
        }
    }

#pragma unroll
    for (int mt = 0; mt < 8; ++mt)
#pragma unroll
        for (int rr = 0; rr < 8; ++rr)
            Out[(size_t)(mt * 16 + rr + hi * 8) * N + n0 + r] = acc[mt][rr];
}

// ---------------- TDM: DMA a [rows x cols] 2-byte-element row-major tile into LDS ----------------
// Descriptor layout per CDNA5 ISA ch.7/8 (async_tensor): group0 = {count/lds/global/type},
// group1 = {data_size, tensor dims, tile dims, strides}, groups 2/3 unused for 2D.
// This toolchain's builtin takes 6 args (extra i32x8 group before cpol) — pass zeros there.
__device__ __forceinline__ void tdm_load_tile_to_lds(const void* gptr, unsigned lds_off,
                                                     unsigned rows, unsigned cols) {
#if __has_builtin(__builtin_amdgcn_tensor_load_to_lds)
    unsigned long long ga = (unsigned long long)(uintptr_t)gptr;
    u32x4 g0;
    g0[0] = 1u;                                   // count=1, user mode
    g0[1] = lds_off;                              // lds_addr (bytes)
    g0[2] = (unsigned)(ga & 0xffffffffu);         // global_addr[31:0]
    g0[3] = (unsigned)((ga >> 32) & 0x01ffffffu)  // global_addr[56:32]
          | (2u << 30);                           // type = 2 ("image")
    i32x8 g1;
    g1[0] = (int)(1u << 16);                      // data_size = 1 (2 bytes/elem)
    g1[1] = (int)((cols & 0xffffu) << 16);        // tensor_dim0[15:0] in bits[63:48]
    g1[2] = (int)((cols >> 16) | ((rows & 0xffffu) << 16)); // dim0[31:16] | dim1[15:0]
    g1[3] = (int)((rows >> 16) | (cols << 16));   // dim1[31:16] | tile_dim0
    g1[4] = (int)(rows & 0xffffu);                // tile_dim1 (tile_dim2 = 0)
    g1[5] = (int)cols;                            // tensor_dim0_stride[31:0]
    g1[6] = 0;                                    // dim0_stride[47:32] | dim1_stride lo
    g1[7] = 0;
    i32x4 gz4 = {0, 0, 0, 0};
    i32x8 gz8 = {0, 0, 0, 0, 0, 0, 0, 0};
    __builtin_amdgcn_tensor_load_to_lds(g0, g1, gz4, gz4, gz8, 0);
#else
    (void)gptr; (void)lds_off; (void)rows; (void)cols;
#endif
}

// ---------------- WMMA projection GEMM with LDS-staged A (8 waves share A) ----------------
// Out[128 x N] = A[128 x K] * W[N x K]^T. A staged to LDS once per block via TDM.
__global__ __launch_bounds__(256) void gemm_proj_lds_kernel(
    const unsigned short* __restrict__ A, const unsigned short* __restrict__ W, int K,
    float* __restrict__ Out, int N)
{
    extern __shared__ unsigned short As[]; // [128 x K] bf16
    const int wave = threadIdx.x >> 5;
    const int lane = threadIdx.x & 31;
    const int r    = lane & 15;
    const int hi   = lane >> 4;
    const int strip = blockIdx.x * 8 + wave;
    const int nstrips = N >> 4;
    const int n0 = strip << 4;

    // ---- stage A into LDS ----
#if __has_builtin(__builtin_amdgcn_tensor_load_to_lds)
    if (threadIdx.x < 32) {                       // one TDM issue per workgroup (wave 0)
        tdm_load_tile_to_lds(A, 0u, 128u, (unsigned)K);
        __builtin_amdgcn_s_wait_tensorcnt(0);
    }
#else
    {
        const int nvec = (128 * K) >> 3;          // uint4 = 8 bf16
        for (int i = threadIdx.x; i < nvec; i += 256)
            ((uint4*)As)[i] = ((const uint4*)A)[i];
    }
#endif
    __syncthreads();

    if (strip < nstrips) {
        v8f acc[8];
#pragma unroll
        for (int i = 0; i < 8; ++i)
#pragma unroll
            for (int j = 0; j < 8; ++j) acc[i][j] = 0.0f;

        for (int k0 = 0; k0 < K; k0 += 32) {
            Frag bfr;
            const unsigned short* wp = W + (size_t)(n0 + r) * K + k0 + hi * 8;
            bfr.u[0] = *(const uint4*)(wp);
            bfr.u[1] = *(const uint4*)(wp + 16);
#pragma unroll
            for (int mt = 0; mt < 8; ++mt) {
                Frag afr;
                const unsigned short* ap = As + (size_t)(mt * 16 + r) * K + k0 + hi * 8;
                afr.u[0] = *(const uint4*)(ap);
                afr.u[1] = *(const uint4*)(ap + 16);
                acc[mt] = __builtin_amdgcn_wmma_f32_16x16x32_bf16(
                    false, afr.v, false, bfr.v, (short)0, acc[mt], false, false);
            }
        }

#pragma unroll
        for (int mt = 0; mt < 8; ++mt)
#pragma unroll
            for (int rr = 0; rr < 8; ++rr)
                Out[(size_t)(mt * 16 + rr + hi * 8) * N + n0 + r] = acc[mt][rr];
    }
}

// ---------------- LSTM gate nonlinearity ----------------
__global__ void lstm_act_kernel(const float* __restrict__ g,
                                const float* __restrict__ bih,
                                const float* __restrict__ bhh,
                                float* __restrict__ c,
                                unsigned short* __restrict__ hout,
                                unsigned short* __restrict__ hseq)
{
    int idx = blockIdx.x * blockDim.x + threadIdx.x;
    if (idx >= BSZ * HH) return;
    int b = idx / HH;
    int j = idx - b * HH;
    const float* gr = g + (size_t)b * H4;
    float gi = gr[j]          + bih[j]          + bhh[j];
    float gf = gr[HH + j]     + bih[HH + j]     + bhh[HH + j];
    float gg = gr[2 * HH + j] + bih[2 * HH + j] + bhh[2 * HH + j];
    float go = gr[3 * HH + j] + bih[3 * HH + j] + bhh[3 * HH + j];
    float si = 1.0f / (1.0f + __expf(-gi));
    float sf = 1.0f / (1.0f + __expf(-gf));
    float so = 1.0f / (1.0f + __expf(-go));
    float cc = sf * c[idx] + si * tanhf(gg);
    c[idx] = cc;
    float hh = so * tanhf(cc);
    unsigned short hb = f32_to_bf16(hh);
    hout[(size_t)b * HP + j] = hb;
    if (hseq) hseq[(size_t)b * HP + j] = hb;
}

// ---------------- softmax cross-entropy (per decoder step) ----------------
__global__ __launch_bounds__(256) void softmax_loss_kernel(
    const float* __restrict__ logits, const float* __restrict__ bout,
    const int* __restrict__ tgt, float* __restrict__ loss_out)
{
    __shared__ float red[256];
    const int b   = blockIdx.x;
    const int tid = threadIdx.x;
    const float* row = logits + (size_t)b * VV;

    float mx = -3.4e38f;
    for (int v = tid; v < VV; v += 256) mx = fmaxf(mx, row[v] + bout[v]);
    red[tid] = mx;
    __syncthreads();
    for (int s = 128; s > 0; s >>= 1) {
        if (tid < s) red[tid] = fmaxf(red[tid], red[tid + s]);
        __syncthreads();
    }
    mx = red[0];
    __syncthreads();

    float se = 0.0f;
    for (int v = tid; v < VV; v += 256) se += __expf(row[v] + bout[v] - mx);
    red[tid] = se;
    __syncthreads();
    for (int s = 128; s > 0; s >>= 1) {
        if (tid < s) red[tid] += red[tid + s];
        __syncthreads();
    }
    if (tid == 0) {
        int tv = tgt[b];
        float lse = mx + __logf(red[0]);
        float tl  = row[tv] + bout[tv];
        atomicAdd(loss_out, (lse - tl) * (1.0f / (float)BSZ));
    }
}

// ---------------- host side ----------------
extern "C" void kernel_launch(void* const* d_in, const int* in_sizes, int n_in,
                              void* d_out, int out_size, void* d_ws, size_t ws_size,
                              hipStream_t stream) {
    (void)in_sizes; (void)n_in; (void)out_size; (void)ws_size;

    const int*   x        = (const int*)  d_in[0];
    const int*   y        = (const int*)  d_in[1];
    const float* enc_emb  = (const float*)d_in[2];
    const float* dec_emb  = (const float*)d_in[3];
    const float* e_Wih1   = (const float*)d_in[4];
    const float* e_Whh1   = (const float*)d_in[5];
    const float* e_bih1   = (const float*)d_in[6];
    const float* e_bhh1   = (const float*)d_in[7];
    const float* e_Wih2   = (const float*)d_in[8];
    const float* e_Whh2   = (const float*)d_in[9];
    const float* e_bih2   = (const float*)d_in[10];
    const float* e_bhh2   = (const float*)d_in[11];
    const float* d_Wih1   = (const float*)d_in[12];
    const float* d_Whh1   = (const float*)d_in[13];
    const float* d_bih1   = (const float*)d_in[14];
    const float* d_bhh1   = (const float*)d_in[15];
    const float* d_Wih2   = (const float*)d_in[16];
    const float* d_Whh2   = (const float*)d_in[17];
    const float* d_bih2   = (const float*)d_in[18];
    const float* d_bhh2   = (const float*)d_in[19];
    const float* Wout     = (const float*)d_in[20];
    const float* bout     = (const float*)d_in[21];
    float* loss = (float*)d_out;

    // workspace layout
    char* ws = (char*)d_ws;
    size_t off = 0;
    auto alloc = [&](size_t bytes) -> void* {
        void* p = ws + off;
        off = (off + bytes + 255) & ~(size_t)255;
        return p;
    };
    unsigned short* ex     = (unsigned short*)alloc((size_t)SXX * BSZ * EE * 2);
    unsigned short* ey     = (unsigned short*)alloc((size_t)TDD * BSZ * EE * 2);
    unsigned short* eWih1b = (unsigned short*)alloc((size_t)H4 * EE * 2);
    unsigned short* eWhh1b = (unsigned short*)alloc((size_t)H4 * HP * 2);
    unsigned short* eWih2b = (unsigned short*)alloc((size_t)H4 * HP * 2);
    unsigned short* eWhh2b = (unsigned short*)alloc((size_t)H4 * HP * 2);
    unsigned short* dWih1b = (unsigned short*)alloc((size_t)H4 * EE * 2);
    unsigned short* dWhh1b = (unsigned short*)alloc((size_t)H4 * HP * 2);
    unsigned short* dWih2b = (unsigned short*)alloc((size_t)H4 * HP * 2);
    unsigned short* dWhh2b = (unsigned short*)alloc((size_t)H4 * HP * 2);
    unsigned short* Woutb  = (unsigned short*)alloc((size_t)VV * HP * 2);
    unsigned short* seq1   = (unsigned short*)alloc((size_t)SXX * BSZ * HP * 2);
    unsigned short* h1buf  = (unsigned short*)alloc((size_t)BSZ * HP * 2);
    unsigned short* h2buf  = (unsigned short*)alloc((size_t)BSZ * HP * 2);
    float* c1     = (float*)alloc((size_t)BSZ * HH * 4);
    float* c2     = (float*)alloc((size_t)BSZ * HH * 4);
    float* g      = (float*)alloc((size_t)BSZ * H4 * 4);
    float* logits = (float*)alloc((size_t)BSZ * VV * 4);

    // ---- init (state/pad zeroing; runs every call -> deterministic) ----
    zero_f32_kernel<<<1, 64, 0, stream>>>(loss, 1);
    zero_u16_kernel<<<512, 256, 0, stream>>>(h1buf, (long)BSZ * HP);
    zero_u16_kernel<<<512, 256, 0, stream>>>(h2buf, (long)BSZ * HP);
    zero_u16_kernel<<<2048, 256, 0, stream>>>(seq1, (long)SXX * BSZ * HP);
    zero_f32_kernel<<<512, 256, 0, stream>>>(c1, (long)BSZ * HH);
    zero_f32_kernel<<<512, 256, 0, stream>>>(c2, (long)BSZ * HH);

    // ---- weight conversion to padded bf16 ----
    auto cvt = [&](const float* src, unsigned short* dst, int N, int K, int Kpad) {
        int total = N * Kpad;
        cvt_pad_kernel<<<(total + 255) / 256, 256, 0, stream>>>(src, dst, N, K, Kpad);
    };
    cvt(e_Wih1, eWih1b, H4, EE, EE);
    cvt(e_Whh1, eWhh1b, H4, HH, HP);
    cvt(e_Wih2, eWih2b, H4, HH, HP);
    cvt(e_Whh2, eWhh2b, H4, HH, HP);
    cvt(d_Wih1, dWih1b, H4, EE, EE);
    cvt(d_Whh1, dWhh1b, H4, HH, HP);
    cvt(d_Wih2, dWih2b, H4, HH, HP);
    cvt(d_Whh2, dWhh2b, H4, HH, HP);
    cvt(Wout,   Woutb,  VV, HH, HP);

    // ---- embeddings ----
    embed_kernel<<<(SXX * BSZ * EE + 255) / 256, 256, 0, stream>>>(x, enc_emb, ex, SXX);
    embed_kernel<<<(TDD * BSZ * EE + 255) / 256, 256, 0, stream>>>(y, dec_emb, ey, TDD);

    auto gemm = [&](const unsigned short* A1, const unsigned short* W1, int K1,
                    const unsigned short* A2, const unsigned short* W2, int K2,
                    float* Out, int N) {
        gemm_nt_128_kernel<<<N / 16, 32, 0, stream>>>(A1, W1, K1, A2, W2, K2, Out, N);
    };
    auto act = [&](const float* bih, const float* bhh, float* c,
                   unsigned short* hout, unsigned short* hseq) {
        lstm_act_kernel<<<(BSZ * HH + 255) / 256, 256, 0, stream>>>(g, bih, bhh, c, hout, hseq);
    };

    const size_t proj_lds = (size_t)BSZ * HP * 2; // 139,264 B of dynamic LDS (fits 320KB WGP)

    // ---- encoder (layers interleaved per timestep; equivalent to layer-major scan) ----
    for (int t = 0; t < SXX; ++t) {
        gemm(ex + (size_t)t * BSZ * EE, eWih1b, EE, h1buf, eWhh1b, HP, g, H4);
        act(e_bih1, e_bhh1, c1, h1buf, seq1 + (size_t)t * BSZ * HP);
        gemm(seq1 + (size_t)t * BSZ * HP, eWih2b, HP, h2buf, eWhh2b, HP, g, H4);
        act(e_bih2, e_bhh2, c2, h2buf, nullptr);
    }

    // ---- decoder (states h1/c1, h2/c2 carry over from encoder) ----
    for (int t = 0; t < TDD; ++t) {
        gemm(ey + (size_t)t * BSZ * EE, dWih1b, EE, h1buf, dWhh1b, HP, g, H4);
        act(d_bih1, d_bhh1, c1, h1buf, nullptr);
        gemm(h1buf, dWih2b, HP, h2buf, dWhh2b, HP, g, H4);
        act(d_bih2, d_bhh2, c2, h2buf, nullptr);
        // projection: 2000 N-strips, 8 per block sharing TDM-staged A in LDS
        gemm_proj_lds_kernel<<<(VV / 16 + 7) / 8, 256, proj_lds, stream>>>(
            h2buf, Woutb, HP, logits, VV);
        softmax_loss_kernel<<<BSZ, 256, 0, stream>>>(logits, bout, y + (size_t)(t + 1) * BSZ, loss);
    }
}
